// cycle_attention_60696477827141
// MI455X (gfx1250) — compile-verified, matrix operand order checked by reading, not run on hardware
//
#include <hip/hip_runtime.h>
#include <hip/hip_bf16.h>

// ---------------------------------------------------------------------------
// cycle-attention for MI455X (gfx1250, wave32, WMMA + TDM)
// Algebraic refactor removes the 17-GFLOP k/v projections:
//   attn = (q@Wk) @ feat   ;   plan@v = (plan@feat^T) @ Wv^T
// Skinny GEMMs use v_wmma_f32_16x16x32_f16. Sinkhorn (300 iters) runs fully
// in LDS. The logits GEMM stages feat tiles through LDS with the Tensor Data
// Mover (tensor_load_to_lds, double buffered, s_wait_tensorcnt), using TDM's
// pad feature (1 dword per 256) to de-conflict the LDS banks.
// A-fragments are loaded per K-step (not cached across the loop) to keep VGPR
// pressure below the spill threshold in the WMMA inner loops.
// ---------------------------------------------------------------------------

typedef _Float16 v16h __attribute__((ext_vector_type(16)));
typedef _Float16 v8h  __attribute__((ext_vector_type(8)));
typedef float    v8f  __attribute__((ext_vector_type(8)));
typedef unsigned int u32x4 __attribute__((ext_vector_type(4)));
typedef int          i32x4 __attribute__((ext_vector_type(4)));
typedef int          i32x8 __attribute__((ext_vector_type(8)));

#define NB    8
#define NC    256
#define NHW   4096
#define NA    5
#define NTHR  256
#define SCALEQ 0.0625f   // 256^-0.5

#define TDM_ROWSTRIDE 257          // 256 + 1 pad dword per row (bank spread)
#define TDM_TILE_F    (32 * TDM_ROWSTRIDE)

#if __has_builtin(__builtin_amdgcn_tensor_load_to_lds)
#define USE_TDM 1
#else
#define USE_TDM 0
#endif

// ---- WMMA helpers (ISA 7.12.2 wave32 layouts) -----------------------------

__device__ __forceinline__ v8f wmma_f16(v16h a, v16h b, v8f c) {
  return __builtin_amdgcn_wmma_f32_16x16x32_f16(false, a, false, b,
                                                (short)0, c, false, false);
}

// K index inside a 32-wide K step for element e of a 16-half fragment.
// lanes 0-15: K = e(0..7), 16+e(8..15); lanes 16-31: +8.
__device__ __forceinline__ int kpat(int e, int hi) {
  return (e < 8 ? e : e + 8) + (hi ? 8 : 0);
}

// A-fragment from a contiguous f16 row (global or LDS), row-major [M][256].
__device__ __forceinline__ v16h load_afrag_f16row(const _Float16* row, int kk, int hi) {
  const v8h* p = (const v8h*)(row + kk * 32 + hi * 8);
  v8h lo = p[0];
  v8h h2 = p[2];              // +16 halves
  v16h a;
#pragma unroll
  for (int e = 0; e < 8; ++e) { a[e] = lo[e]; a[e + 8] = h2[e]; }
  return a;
}

// A-fragment from an f32 LDS row (plan / attn rows), K step at n-offset nb.
__device__ __forceinline__ v16h load_afrag_f32row(const float* row, int nb, int hi) {
  v16h a;
#pragma unroll
  for (int e = 0; e < 16; ++e) a[e] = (_Float16)row[nb + kpat(e, hi)];
  return a;
}

__device__ __forceinline__ v16h zero_frag() {
  v16h a;
#pragma unroll
  for (int e = 0; e < 16; ++e) a[e] = (_Float16)0.f;
  return a;
}

// B-fragment for logits GEMM from global: B[k][n] = feat[(kk*32+k)*NHW + n0+n]
__device__ __forceinline__ v16h load_bfrag_strided(const float* fcol, int kk, int hi) {
  v16h bf;
#pragma unroll
  for (int e = 0; e < 16; ++e)
    bf[e] = (_Float16)fcol[(size_t)(kk * 32 + kpat(e, hi)) * NHW];
  return bf;
}

// B-fragment for logits GEMM from an LDS tile [32][TDM_ROWSTRIDE].
__device__ __forceinline__ v16h load_bfrag_lds(const float* tile, int col, int hi) {
  v16h bf;
#pragma unroll
  for (int e = 0; e < 16; ++e)
    bf[e] = (_Float16)tile[kpat(e, hi) * TDM_ROWSTRIDE + col];
  return bf;
}

// B-fragment for aggregation GEMM: B[k][c] = feat[c*NHW + nb + k]
// base = feat + c*NHW + nb + hi*8  (contiguous: float4 pairs).
__device__ __forceinline__ v16h load_bfrag_rowmajor(const float* base) {
  const float4* q0 = (const float4*)base;
  const float4* q1 = (const float4*)(base + 16);
  float4 x0 = q0[0], x1 = q0[1], y0 = q1[0], y1 = q1[1];
  v16h bf;
  bf[0]=(_Float16)x0.x;  bf[1]=(_Float16)x0.y;  bf[2]=(_Float16)x0.z;  bf[3]=(_Float16)x0.w;
  bf[4]=(_Float16)x1.x;  bf[5]=(_Float16)x1.y;  bf[6]=(_Float16)x1.z;  bf[7]=(_Float16)x1.w;
  bf[8]=(_Float16)y0.x;  bf[9]=(_Float16)y0.y;  bf[10]=(_Float16)y0.z; bf[11]=(_Float16)y0.w;
  bf[12]=(_Float16)y1.x; bf[13]=(_Float16)y1.y; bf[14]=(_Float16)y1.z; bf[15]=(_Float16)y1.w;
  return bf;
}

// ---- Tensor Data Mover: 2D f32 tile -> LDS --------------------------------
// D# per CDNA5 ISA ch.8: group0 {count=1, lds_addr, global_addr, type=2},
// group1 {data_size=4B, pad_enable, pad_interval=256dw, pad_amount=1dw,
//         tensor_dim0/1, tile_dim0/1, tensor_dim0_stride}.
// amdgpu-toolchain (clang-23) builtin: 6 args
//   (uint32x4 g0, int32x8 g1, int32x4 g2, int32x4 g3, int32x8 g4, i32 cpol)
#if USE_TDM
__device__ __forceinline__ unsigned lds_addr_of(const void* p) {
  // generic LDS address: low 32 bits are the LDS byte offset (aperture map)
  return (unsigned)(uintptr_t)p;
}

__device__ __forceinline__ void tdm_load_tile_f32(const float* gsrc, unsigned lds_byte,
                                                  int tile_n /*dim0*/, int tile_k /*dim1*/,
                                                  int tensor_n, int tensor_k,
                                                  int row_stride_elems) {
  unsigned long long ga = (unsigned long long)(uintptr_t)gsrc;
  u32x4 g0 = {
    1u,                                                    // count=1 (user D#)
    lds_byte,                                              // lds_addr
    (unsigned)(ga & 0xFFFFFFFFu),                          // global_addr[31:0]
    ((unsigned)(ga >> 32) & 0x01FFFFFFu) | 0x80000000u     // ga[56:32] | type=2
  };
  i32x8 g1 = {
    (int)((2u << 16) | (1u << 20) | (7u << 22)),           // 4B, pad_en, every 256dw (+1dw)
    (int)(((unsigned)tensor_n & 0xFFFFu) << 16),           // tensor_dim0[15:0]
    (int)((((unsigned)tensor_n >> 16) & 0xFFFFu) |
          (((unsigned)tensor_k & 0xFFFFu) << 16)),         // dim0 hi | dim1 lo
    (int)((((unsigned)tensor_k >> 16) & 0xFFFFu) |
          (((unsigned)tile_n & 0xFFFFu) << 16)),           // dim1 hi | tile_dim0
    (int)((unsigned)tile_k & 0xFFFFu),                     // tile_dim1 (tile_dim2=0)
    (int)(unsigned)row_stride_elems,                       // tensor_dim0_stride[31:0]
    0, 0
  };
  i32x4 gz4 = {0, 0, 0, 0};
  i32x8 gz8 = {0, 0, 0, 0, 0, 0, 0, 0};
  __builtin_amdgcn_tensor_load_to_lds(g0, g1, gz4, gz4, gz8, 0);
}
#endif

// ---- block reductions (256 threads) ---------------------------------------

__device__ __forceinline__ float bred_sum(float v, float* red) {
  const int t = threadIdx.x;
  red[t] = v; __syncthreads();
  for (int s = NTHR / 2; s > 0; s >>= 1) {
    if (t < s) red[t] += red[t + s];
    __syncthreads();
  }
  float r = red[0]; __syncthreads();
  return r;
}

__device__ __forceinline__ float bred_max(float v, float* red) {
  const int t = threadIdx.x;
  red[t] = v; __syncthreads();
  for (int s = NTHR / 2; s > 0; s >>= 1) {
    if (t < s) red[t] = fmaxf(red[t], red[t + s]);
    __syncthreads();
  }
  float r = red[0]; __syncthreads();
  return r;
}

// ===========================================================================
// K0: per (branch,b): masked-mean prototype + part tokens, q = protoA@Wq^T,
//     qk = q@Wk, pack qk as f16 [16][256] (rows 5..15 zero) for WMMA A-frags.
// ===========================================================================
__global__ void k0_proto(const float* __restrict__ sf1, const float* __restrict__ sf2,
                         const int* __restrict__ sm1, const int* __restrict__ sm2,
                         const float* __restrict__ Wq, const float* __restrict__ Wk,
                         const float* __restrict__ tok,
                         float* __restrict__ protoA_ws, _Float16* __restrict__ qkh_ws) {
  const int bidx = blockIdx.x, br = bidx >> 3, b = bidx & 7, tid = threadIdx.x;
  const float* feat = (br ? sf2 : sf1) + (size_t)b * NC * NHW;
  const int*   mask = (br ? sm2 : sm1) + (size_t)b * NHW;

  __shared__ float msh[NHW];
  __shared__ float red[NTHR];
  __shared__ float mean[NC];
  __shared__ float protoSh[NA * NC];
  __shared__ float qSh[NA * NC];

  float cnt = 0.f;
  for (int n = tid; n < NHW; n += NTHR) {
    float m = (mask[n] == 1) ? 1.f : 0.f;
    msh[n] = m; cnt += m;
  }
  float nfg = bred_sum(cnt, red);     // syncs internally

  // masked mean, one channel per thread
  {
    const float* row = feat + (size_t)tid * NHW;
    float s = 0.f;
    for (int n = 0; n < NHW; ++n) s += row[n] * msh[n];
    mean[tid] = s / (nfg + 1e-5f);
  }
  __syncthreads();

  float* protoA = protoA_ws + (size_t)bidx * NA * NC;
#pragma unroll
  for (int a = 0; a < NA; ++a) {
    float v = mean[tid] + tok[a * NC + tid];
    protoSh[a * NC + tid] = v;
    protoA[a * NC + tid]  = v;
  }
  __syncthreads();

  // q[a][c] = sum_j protoA[a][j] * Wq[c][j]
#pragma unroll
  for (int a = 0; a < NA; ++a) {
    const float* w = Wq + (size_t)tid * NC;
    const float* p = protoSh + a * NC;
    float s = 0.f;
    for (int j = 0; j < NC; ++j) s += p[j] * w[j];
    qSh[a * NC + tid] = s;
  }
  __syncthreads();

  // qk[a][j] = sum_c q[a][c] * Wk[c][j]   (thread owns column j = tid)
  float qk[NA] = {0.f, 0.f, 0.f, 0.f, 0.f};
  for (int c = 0; c < NC; ++c) {
    float w = Wk[(size_t)c * NC + tid];
#pragma unroll
    for (int a = 0; a < NA; ++a) qk[a] += qSh[a * NC + c] * w;
  }
  _Float16* dst = qkh_ws + (size_t)bidx * 16 * NC;
#pragma unroll
  for (int a = 0; a < NA; ++a) dst[a * NC + tid] = (_Float16)qk[a];
#pragma unroll
  for (int r = NA; r < 16; ++r) dst[r * NC + tid] = (_Float16)0.f;
}

// ===========================================================================
// K1: per (branch,b) fused: TDM-staged WMMA logits -> masked softmax ->
//     Sinkhorn(300) in LDS -> transport plan -> WMMA aggregation -> @Wv^T
//     + proto. Writes mem rows [branch*5 .. branch*5+4].
// Dynamic LDS: K[5][4096] + v + mask + reductions + 2 TDM tiles ~ 187KB.
// ===========================================================================
__global__ void k1_sinkhorn(const float* __restrict__ sf1, const float* __restrict__ sf2,
                            const int* __restrict__ sm1, const int* __restrict__ sm2,
                            const float* __restrict__ Wv, float* __restrict__ out_mem,
                            const float* __restrict__ protoA_ws, float* __restrict__ pout_ws,
                            const _Float16* __restrict__ qkh_ws) {
  const int bidx = blockIdx.x, br = bidx >> 3, b = bidx & 7;
  const int tid = threadIdx.x, lane = tid & 31, wave = tid >> 5, hi = (lane >= 16);
  const float* feat = (br ? sf2 : sf1) + (size_t)b * NC * NHW;
  const int*   mask = (br ? sm2 : sm1) + (size_t)b * NHW;
  const _Float16* qkh = qkh_ws + (size_t)bidx * 16 * NC;
  const float* protoA = protoA_ws + (size_t)bidx * NA * NC;
  float* pout = pout_ws + (size_t)bidx * NA * NC;

  extern __shared__ float sm[];
  float* Kp   = sm;                   // 5*4096 : logits -> attn -> K -> plan
  float* vv   = Kp + NA * NHW;        // 4096
  float* msh  = vv + NHW;             // 4096
  float* redA = msh + NHW;            // 5*256
  float* psb  = redA + NA * NTHR;     // 5*256
  float* red  = psb + NA * NC;        // 256
  float* scal = red + NTHR;           // 16 : u[0..4]
  float* tbuf = scal + 16;            // 2 * 32*257 : TDM double buffer

  // stage mask, count foreground
  float cnt = 0.f;
  for (int n = tid; n < NHW; n += NTHR) {
    float m = (mask[n] == 1) ? 1.f : 0.f;
    msh[n] = m; cnt += m;
  }
  const float nfg  = bred_sum(cnt, red);
  const float bden = 1.f / fmaxf(nfg, 1e-30f);   // b[n] = m[n]/nfg

  // ---- logits[a][n] = sum_k qk[a][k] * feat[k][n]  (WMMA, M=16,K=256) ----
  const _Float16* qrow = qkh + (size_t)(lane & 15) * NC;  // A row for this lane

#if USE_TDM
  // TDM-staged: per 256-wide n-block, DMA feat[kk*32..+32)[nblk*256..+256)
  // into LDS (double buffered), 8 waves x two 16-wide n-subtiles each.
  for (int nblk = 0; nblk < NHW / 256; ++nblk) {
    if (wave == 0) {
      tdm_load_tile_f32(feat + (size_t)nblk * 256, lds_addr_of(tbuf),
                        256, 32, NHW, NC, NHW);
      __builtin_amdgcn_s_wait_tensorcnt((short)0);
    }
    __syncthreads();
    v8f accs[2] = {{}, {}};
    for (int kk = 0; kk < 8; ++kk) {
      const float* cur = tbuf + (kk & 1) * TDM_TILE_F;
      if (wave == 0 && kk < 7) {
        tdm_load_tile_f32(feat + (size_t)(kk + 1) * 32 * NHW + (size_t)nblk * 256,
                          lds_addr_of(tbuf + ((kk + 1) & 1) * TDM_TILE_F),
                          256, 32, NHW, NC, NHW);
      }
      const v16h af = load_afrag_f16row(qrow, kk, hi);   // L1-cached, no spill
#pragma unroll
      for (int tt = 0; tt < 2; ++tt) {
        const int col = (wave * 2 + tt) * 16 + (lane & 15);
        accs[tt] = wmma_f16(af, load_bfrag_lds(cur, col, hi), accs[tt]);
      }
      if (wave == 0 && kk < 7) __builtin_amdgcn_s_wait_tensorcnt((short)0);
      __syncthreads();
    }
    if (lane < 16) {
#pragma unroll
      for (int tt = 0; tt < 2; ++tt) {
        const int n0 = nblk * 256 + (wave * 2 + tt) * 16;
#pragma unroll
        for (int r = 0; r < NA; ++r) Kp[r * NHW + n0 + lane] = accs[tt][r];
      }
    }
  }
#else
  (void)tbuf;
  for (int t = wave; t < NHW / 16; t += 8) {
    const int n0 = t * 16;
    const float* fcol = feat + n0 + (lane & 15);
    v8f acc = {};
#pragma unroll 1
    for (int kk = 0; kk < 8; ++kk)
      acc = wmma_f16(load_afrag_f16row(qrow, kk, hi),
                     load_bfrag_strided(fcol, kk, hi), acc);
    if (lane < 16) {
#pragma unroll
      for (int r = 0; r < NA; ++r) Kp[r * NHW + n0 + lane] = acc[r];
    }
  }
#endif
  __syncthreads();

  // ---- masked softmax per agent row, then Sinkhorn kernel K=exp(2a-2) ----
  for (int a = 0; a < NA; ++a) {
    float lmax = -3.4e38f;
    for (int n = tid; n < NHW; n += NTHR)
      if (msh[n] > 0.f) lmax = fmaxf(lmax, Kp[a * NHW + n]);
    lmax = bred_max(lmax, red);
    float ls = 0.f;
    for (int n = tid; n < NHW; n += NTHR)
      if (msh[n] > 0.f) ls += expf(Kp[a * NHW + n] - lmax);
    ls = bred_sum(ls, red);
    const float invZ = 1.f / fmaxf(ls, 1e-30f);
    for (int n = tid; n < NHW; n += NTHR) {
      if (msh[n] > 0.f) {
        float att = expf(Kp[a * NHW + n] - lmax) * invZ;
        Kp[a * NHW + n] = expf(2.f * att - 2.f);   // exp(-(1-att)/0.5)
      } else {
        Kp[a * NHW + n] = 0.f;
      }
    }
    __syncthreads();
  }

  // ---- Sinkhorn: u=a=1/5; 300x { v=b/(K^T u); u=a/(K v) }; final v ----
  if (tid < NA) scal[tid] = 1.f / (float)NA;
  __syncthreads();
  for (int it = 0; it < 301; ++it) {
    const float u0 = scal[0], u1 = scal[1], u2 = scal[2], u3 = scal[3], u4 = scal[4];
    for (int n = tid; n < NHW; n += NTHR) {
      float s = u0 * Kp[n] + u1 * Kp[NHW + n] + u2 * Kp[2 * NHW + n]
              + u3 * Kp[3 * NHW + n] + u4 * Kp[4 * NHW + n];
      vv[n] = (msh[n] * bden) / fmaxf(s, 1e-30f);
    }
    __syncthreads();
    if (it == 300) break;
    float acc[NA] = {0.f, 0.f, 0.f, 0.f, 0.f};
    for (int n = tid; n < NHW; n += NTHR) {
      float vn = vv[n];
#pragma unroll
      for (int a = 0; a < NA; ++a) acc[a] += Kp[a * NHW + n] * vn;
    }
#pragma unroll
    for (int a = 0; a < NA; ++a) redA[a * NTHR + tid] = acc[a];
    __syncthreads();
    for (int s = NTHR / 2; s > 0; s >>= 1) {
      if (tid < s) {
#pragma unroll
        for (int a = 0; a < NA; ++a) redA[a * NTHR + tid] += redA[a * NTHR + tid + s];
      }
      __syncthreads();
    }
    if (tid < NA) scal[tid] = (1.f / (float)NA) / fmaxf(redA[tid * NTHR], 1e-30f);
    __syncthreads();
  }
  // plan = u * K * v (in place)
  {
    const float u0 = scal[0], u1 = scal[1], u2 = scal[2], u3 = scal[3], u4 = scal[4];
    for (int n = tid; n < NHW; n += NTHR) {
      float vn = vv[n];
      Kp[n]            *= u0 * vn;
      Kp[NHW + n]      *= u1 * vn;
      Kp[2 * NHW + n]  *= u2 * vn;
      Kp[3 * NHW + n]  *= u3 * vn;
      Kp[4 * NHW + n]  *= u4 * vn;
    }
  }
  __syncthreads();

  // ---- ps[a][c] = sum_n plan[a][n] * feat[c][n]  (WMMA, K=4096) ----
  {
    const int t0 = wave * 2;                     // 2 of 16 c-tiles per wave
    const int row = lane & 15;
    v8f accs[2] = {{}, {}};
    for (int kk = 0; kk < NHW / 32; ++kk) {
      const int nb = kk * 32;
      v16h af = (row < NA) ? load_afrag_f32row(Kp + row * NHW, nb, hi) : zero_frag();
#pragma unroll
      for (int tt = 0; tt < 2; ++tt) {
        const int c0 = (t0 + tt) * 16;
        const float* base = feat + (size_t)(c0 + (lane & 15)) * NHW + nb + hi * 8;
        __builtin_prefetch(base + 32, 0, 1);
        accs[tt] = wmma_f16(af, load_bfrag_rowmajor(base), accs[tt]);
      }
    }
    if (lane < 16) {
#pragma unroll
      for (int tt = 0; tt < 2; ++tt) {
        const int c0 = (t0 + tt) * 16;
#pragma unroll
        for (int r = 0; r < NA; ++r) psb[r * NC + c0 + lane] = accs[tt][r];
      }
    }
  }
  __syncthreads();

  // ---- out = protoA + ps @ Wv^T ; write mem rows branch*5..branch*5+4 ----
  {
    const int c = tid;
    const float* wr = Wv + (size_t)c * NC;
#pragma unroll
    for (int a = 0; a < NA; ++a) {
      const float* pr = psb + a * NC;
      float s = 0.f;
      for (int j = 0; j < NC; ++j) s += pr[j] * wr[j];
      float val = protoA[a * NC + c] + s;
      pout[a * NC + c] = val;
      out_mem[((size_t)b * 20 + br * NA + a) * NC + c] = val;
    }
  }
}

// ===========================================================================
// K2: per b: fg thresholding, curr = [p1;p2], WMMA query-attention
//     (logits * SCALE * fg -> softmax -> *fg -> WMMA aggregation),
//     writes mem rows 10..19 and the 20-row mean prototype.
// Dynamic LDS: attn[10][4096] + fg + curr + agg ~ 210KB (< 320KB/WGP).
// ===========================================================================
__global__ void k2_query(const float* __restrict__ qfeat, const float* __restrict__ qpred,
                         const float* __restrict__ pout_ws, float* __restrict__ d_out) {
  const int b = blockIdx.x, tid = threadIdx.x;
  const int lane = tid & 31, wave = tid >> 5, hi = (lane >= 16);
  const float* feat = qfeat + (size_t)b * NC * NHW;

  extern __shared__ float sm[];
  float* Lg   = sm;                        // 10*4096
  float* fg   = Lg + 10 * NHW;             // 4096
  float* cur0 = fg + NHW;                  // 10*256
  float* agg  = cur0 + 10 * NC;            // 10*256
  float* red  = agg + 10 * NC;             // 256
  _Float16* curh = (_Float16*)(red + NTHR);// 16*256 halves

  // fg = threshold(softmax(query_prediction)[1])
  for (int n = tid; n < NHW; n += NTHR) {
    float e0 = qpred[((size_t)b * 2 + 0) * NHW + n];
    float e1 = qpred[((size_t)b * 2 + 1) * NHW + n];
    float m = fmaxf(e0, e1);
    float x0 = expf(e0 - m), x1 = expf(e1 - m);
    float f = x1 / (x0 + x1);
    f = (f > 0.7f) ? 1.f : ((f < 0.3f) ? 0.f : f);
    fg[n] = f;
  }
  // curr0 = concat(p1, p2), f16 copy padded to 16 rows
  {
    const int c = tid;
#pragma unroll
    for (int i = 0; i < 10; ++i) {
      const int brx = i / NA, a = i % NA;
      float v = pout_ws[(((size_t)brx * NB + b) * NA + a) * NC + c];
      cur0[i * NC + c] = v;
      curh[i * NC + c] = (_Float16)v;
    }
#pragma unroll
    for (int i = 10; i < 16; ++i) curh[i * NC + c] = (_Float16)0.f;
  }
  __syncthreads();

  // ---- logits[i][n] = SCALE*fg[n]* sum_c curr[i][c]*feat[c][n] ----
  const _Float16* qrow = curh + (size_t)(lane & 15) * NC;  // A row in LDS
  for (int t = wave; t < NHW / 16; t += 8) {
    const int n0 = t * 16;
    const float* fcol = feat + n0 + (lane & 15);
    __builtin_prefetch(fcol + 16, 0, 1);
    v8f acc = {};
#pragma unroll 1
    for (int kk = 0; kk < 8; ++kk)
      acc = wmma_f16(load_afrag_f16row(qrow, kk, hi),
                     load_bfrag_strided(fcol, kk, hi), acc);
    if (lane < 16) {                       // rows 0..7
      const int n = n0 + lane;
      const float g = fg[n] * SCALEQ;
#pragma unroll
      for (int r = 0; r < 8; ++r) Lg[r * NHW + n] = acc[r] * g;
    } else {                               // rows 8..9
      const int n = n0 + lane - 16;
      const float g = fg[n] * SCALEQ;
#pragma unroll
      for (int r = 0; r < 2; ++r) Lg[(8 + r) * NHW + n] = acc[r] * g;
    }
  }
  __syncthreads();

  // ---- softmax over n per row, then fold fg for aggregation ----
  for (int i = 0; i < 10; ++i) {
    float lmax = -3.4e38f;
    for (int n = tid; n < NHW; n += NTHR) lmax = fmaxf(lmax, Lg[i * NHW + n]);
    lmax = bred_max(lmax, red);
    float ls = 0.f;
    for (int n = tid; n < NHW; n += NTHR) ls += expf(Lg[i * NHW + n] - lmax);
    ls = bred_sum(ls, red);
    const float invZ = 1.f / fmaxf(ls, 1e-30f);
    for (int n = tid; n < NHW; n += NTHR)
      Lg[i * NHW + n] = expf(Lg[i * NHW + n] - lmax) * invZ * fg[n];
    __syncthreads();
  }

  // ---- agg[i][c] = sum_n attn*fg * feat[c][n]  (WMMA, K=4096) ----
  {
    const int t0 = wave * 2;
    const int row = lane & 15;
    v8f accs[2] = {{}, {}};
    for (int kk = 0; kk < NHW / 32; ++kk) {
      const int nb = kk * 32;
      v16h af = (row < 10) ? load_afrag_f32row(Lg + row * NHW, nb, hi) : zero_frag();
#pragma unroll
      for (int tt = 0; tt < 2; ++tt) {
        const int c0 = (t0 + tt) * 16;
        const float* base = feat + (size_t)(c0 + (lane & 15)) * NHW + nb + hi * 8;
        __builtin_prefetch(base + 32, 0, 1);
        accs[tt] = wmma_f16(af, load_bfrag_rowmajor(base), accs[tt]);
      }
    }
#pragma unroll
    for (int tt = 0; tt < 2; ++tt) {
      const int c0 = (t0 + tt) * 16;
      if (lane < 16) {
#pragma unroll
        for (int r = 0; r < 8; ++r) agg[r * NC + c0 + lane] = accs[tt][r];
      } else {
#pragma unroll
        for (int r = 0; r < 2; ++r) agg[(8 + r) * NC + c0 + lane - 16] = accs[tt][r];
      }
    }
  }
  __syncthreads();

  // ---- mem rows 10..19 and curr_prototype = mean over 20 rows ----
  {
    const int c = tid;
    float s20 = 0.f;
#pragma unroll
    for (int i = 0; i < 10; ++i) {
      float cv = cur0[i * NC + c];
      float nv = cv + agg[i * NC + c];
      d_out[((size_t)b * 20 + 10 + i) * NC + c] = nv;
      s20 += cv + nv;
    }
    d_out[(size_t)NB * 20 * NC + (size_t)b * NC + c] = s20 * (1.f / 20.f);
  }
}

// ===========================================================================

extern "C" void kernel_launch(void* const* d_in, const int* in_sizes, int n_in,
                              void* d_out, int out_size, void* d_ws, size_t ws_size,
                              hipStream_t stream) {
  (void)in_sizes; (void)n_in; (void)out_size; (void)ws_size;
  const float* sf1 = (const float*)d_in[0];
  const float* sf2 = (const float*)d_in[1];
  const int*   sm1 = (const int*)d_in[2];
  const int*   sm2 = (const int*)d_in[3];
  const float* qf  = (const float*)d_in[4];
  const float* qp  = (const float*)d_in[5];
  const float* Wq  = (const float*)d_in[6];
  const float* Wk  = (const float*)d_in[7];
  const float* Wv  = (const float*)d_in[8];
  const float* tok = (const float*)d_in[9];
  float* out = (float*)d_out;

  // workspace: protoA[16][5][256] f32 | pout[16][5][256] f32 | qk f16 [16][16][256]
  float* wsf = (float*)d_ws;
  float* protoA = wsf;                        // 20480 f32
  float* pout   = wsf + 20480;                // 20480 f32
  _Float16* qkh = (_Float16*)(wsf + 40960);   // 65536 halves

  const size_t smem1 = (size_t)(NA * NHW + NHW + NHW + NA * NTHR + NA * NC + NTHR + 16
                                + 2 * TDM_TILE_F) * sizeof(float);
  const size_t smem2 = (size_t)(10 * NHW + NHW + 10 * NC + 10 * NC + NTHR) * sizeof(float)
                     + (size_t)16 * NC * sizeof(_Float16);

  (void)hipFuncSetAttribute((const void*)k1_sinkhorn,
                            hipFuncAttributeMaxDynamicSharedMemorySize, (int)smem1);
  (void)hipFuncSetAttribute((const void*)k2_query,
                            hipFuncAttributeMaxDynamicSharedMemorySize, (int)smem2);

  k0_proto   <<<dim3(16), dim3(NTHR), 0,     stream>>>(sf1, sf2, sm1, sm2, Wq, Wk, tok, protoA, qkh);
  k1_sinkhorn<<<dim3(16), dim3(NTHR), smem1, stream>>>(sf1, sf2, sm1, sm2, Wv, out, protoA, pout, qkh);
  k2_query   <<<dim3(8),  dim3(NTHR), smem2, stream>>>(qf, qp, pout, out);
}
